// FireSelfAttention_3496103379319
// MI455X (gfx1250) — matrix operand order repaired and checked by reading, not compiled, not run on hardware
//
#include <hip/hip_runtime.h>

// ---------------------------------------------------------------------------
// FIRE self-attention for MI455X (gfx1250, wave32, WMMA bf16 16x16x32)
// B=8, S=1024, D=512, H=8, KD=64, HID=32
// ---------------------------------------------------------------------------

#define B_ 8
#define S_ 1024
#define D_ 512
#define H_ 8
#define KD_ 64
#define HID_ 32

typedef __bf16 bf16_t;
typedef bf16_t v16bf __attribute__((ext_vector_type(16)));
typedef bf16_t v8bfv __attribute__((ext_vector_type(8)));
typedef float  v8f   __attribute__((ext_vector_type(8)));

__device__ inline v8f wmma_bf16(v16bf a, v16bf b, v8f c) {
  // D = A(16x32 bf16) x B(32x16 bf16) + C(16x16 f32)
  return __builtin_amdgcn_wmma_f32_16x16x32_bf16(
      /*neg_a=*/false, a, /*neg_b=*/false, b,
      /*c_mod=*/(short)0, c, /*reuse_a=*/false, /*reuse_b=*/false);
}

// A-operand (16x32, 16-bit) per-lane gather: lane l (m=l%16, g=l/16) holds
// K = {g*16+0..15} in elems 0..15.
// `row` points at the row's K-chunk base (already offset by the k-step).
__device__ inline v16bf load_A(const bf16_t* __restrict__ row, int g) {
  v8bfv lo = *(const v8bfv*)(row + g * 16);
  v8bfv hi = *(const v8bfv*)(row + g * 16 + 8);
  v16bf a;
#pragma unroll
  for (int e = 0; e < 8; ++e) { a[e] = lo[e]; a[e + 8] = hi[e]; }
  return a;
}

__device__ inline float gelu_erf(float x) {
  return 0.5f * x * (1.0f + erff(x * 0.70710678118654752f));
}

__device__ inline float redmax16(float x) {
#pragma unroll
  for (int m = 1; m < 16; m <<= 1) x = fmaxf(x, __shfl_xor(x, m, 32));
  return x;
}
__device__ inline float redsum16(float x) {
#pragma unroll
  for (int m = 1; m < 16; m <<= 1) x += __shfl_xor(x, m, 32);
  return x;
}

__device__ inline v8f v8f_zero() {
  v8f z = {0.f, 0.f, 0.f, 0.f, 0.f, 0.f, 0.f, 0.f};
  return z;
}

// ---------------------------------------------------------------------------
// f32 -> bf16 conversion
// ---------------------------------------------------------------------------
__global__ void cvt_bf16_kernel(const float* __restrict__ in,
                                bf16_t* __restrict__ out, int n) {
  int i = blockIdx.x * blockDim.x + threadIdx.x;
  if (i < n) out[i] = (bf16_t)in[i];
}

// ---------------------------------------------------------------------------
// QKV projection: q/k -> [B,H,S,KD] bf16, v -> [B,H,KD,S] bf16 (transposed)
// One wave per (b, which, h, 16-row tile). 16x64 output tile, K-loop over D.
// ---------------------------------------------------------------------------
__global__ void qkv_kernel(const bf16_t* __restrict__ src_bf,
                           const bf16_t* __restrict__ wq,
                           const bf16_t* __restrict__ wk,
                           const bf16_t* __restrict__ wv,
                           bf16_t* __restrict__ qb, bf16_t* __restrict__ kb,
                           bf16_t* __restrict__ vtb) {
  int bid = blockIdx.x;
  int it = bid & 63;  bid >>= 6;        // S/16 = 64
  int h  = bid & 7;   bid >>= 3;        // H = 8
  int which = bid % 3;
  int b = bid / 3;

  const bf16_t* W = (which == 0) ? wq : (which == 1) ? wk : wv;
  int l = threadIdx.x, m = l & 15, g = l >> 4;
  int i0 = it * 16;
  const bf16_t* arow = src_bf + ((size_t)b * S_ + i0 + m) * D_;
  // B rows: weight row n = h*64 + t*16 + m (K = d contiguous)
  const bf16_t* brow_base = W + ((size_t)h * KD_ + m) * D_;

  v8f acc[4];
#pragma unroll
  for (int t = 0; t < 4; ++t) acc[t] = v8f_zero();

  for (int kk = 0; kk < D_ / 32; ++kk) {
    int d0 = kk * 32;
    v16bf a = load_A(arow + d0, g);
#pragma unroll
    for (int t = 0; t < 4; ++t) {
      v16bf bm = *(const v16bf*)(brow_base + (size_t)(t * 16) * D_ + d0 + g * 16);
      acc[t] = wmma_bf16(a, bm, acc[t]);
    }
  }

  size_t bh = (size_t)b * H_ + h;
  if (which == 2) {  // V transposed: [B,H,KD,S]
#pragma unroll
    for (int t = 0; t < 4; ++t)
#pragma unroll
      for (int r = 0; r < 8; ++r) {
        int srow = i0 + r + 8 * g;
        int kd = t * 16 + m;
        vtb[(bh * KD_ + kd) * S_ + srow] = (bf16_t)acc[t][r];
      }
  } else {           // Q/K: [B,H,S,KD]
    bf16_t* outp = (which == 0) ? qb : kb;
#pragma unroll
    for (int t = 0; t < 4; ++t)
#pragma unroll
      for (int r = 0; r < 8; ++r) {
        int srow = i0 + r + 8 * g;
        outp[(bh * S_ + srow) * KD_ + t * 16 + m] = (bf16_t)acc[t][r];
      }
  }
}

// ---------------------------------------------------------------------------
// FIRE bias: bias[h,i,j] = f_theta(raw[h,i,j]) for j<=i, -inf for j>i.
// One wave per (h, i, 16-j tile): 16 scalar elements run through the MLP;
// the 32x32 hidden layer is 2 WMMAs (elements are the 16 A-rows).
// ---------------------------------------------------------------------------
__global__ void fire_bias_kernel(const float* __restrict__ c_raw,
                                 const float* __restrict__ Lp,
                                 const float* __restrict__ w1,
                                 const float* __restrict__ b1,
                                 const bf16_t* __restrict__ w2bf,
                                 const float* __restrict__ b2,
                                 const float* __restrict__ w3,
                                 const float* __restrict__ b3,
                                 float* __restrict__ bias) {
  int bid = blockIdx.x;
  int jt = bid & 63;            // S/16
  int i  = (bid >> 6) & 1023;   // S
  int h  = bid >> 16;           // 64*1024 = 2^16 tiles per head
  int j0 = jt * 16;
  int l = threadIdx.x, m = l & 15, g = l >> 4;
  float* brow = bias + ((size_t)h * S_ + i) * S_;

  if (j0 > i) {  // fully masked tile
    if (l < 16) brow[j0 + l] = -__builtin_inff();
    return;
  }

  float c = log1pf(expf(c_raw[h]));  // softplus
  float fi = (float)i;
  float den = log1pf(c * fmaxf(Lp[h], fi + 1.0f));
  int j = j0 + m;                    // this lane's element (row M = m)
  float d = fi - (float)j;
  float raw = (d > 0.0f) ? (log1pf(c * d) / den) : 0.0f;

  // h1 = gelu(raw*w1 + b1), packed directly into A-operand layout
  v16bf a;
#pragma unroll
  for (int e = 0; e < 16; ++e) {
    int k = g * 16 + e;
    float x = raw * w1[h * HID_ + k] + b1[h * HID_ + k];
    a[e] = (bf16_t)gelu_erf(x);
  }

  // hidden layer: h2[n] = sum_k h1[k] * W2[n,k]  -> B[k][n] = W2[n][k]
  const bf16_t* w2h = w2bf + (size_t)h * HID_ * HID_;
  v16bf bw0 = *(const v16bf*)(w2h + (size_t)m * HID_ + g * 16);
  v16bf bw1 = *(const v16bf*)(w2h + (size_t)(16 + m) * HID_ + g * 16);
  v8f c0 = v8f_zero(), c1 = v8f_zero();
  c0 = wmma_bf16(a, bw0, c0);
  c1 = wmma_bf16(a, bw1, c1);

  float w3a = w3[h * HID_ + m],      w3b = w3[h * HID_ + 16 + m];
  float b2a = b2[h * HID_ + m],      b2b = b2[h * HID_ + 16 + m];
  float bb3 = b3[h];

#pragma unroll
  for (int r = 0; r < 8; ++r) {
    float h2a = gelu_erf(c0[r] + b2a);
    float h2b = gelu_erf(c1[r] + b2b);
    float part = redsum16(h2a * w3a + h2b * w3b);  // dot over HID
    if (m == r) {                    // lane r (g=0) -> elem r; lane 16+r -> elem 8+r
      int me = r + 8 * g;
      int jj = j0 + me;
      float val = part + bb3;
      if (jj > i) val = -__builtin_inff();
      brow[jj] = val;
    }
  }
}

// ---------------------------------------------------------------------------
// Flash attention: one wave per (b, h, 16-row i-tile); j-steps of 32 with
// online softmax. QK^T: 4 WMMAs/step, P.V: 4 WMMAs/step. P is moved from the
// C layout to the A layout through LDS.
// ---------------------------------------------------------------------------
__global__ void attn_kernel(const bf16_t* __restrict__ qb,
                            const bf16_t* __restrict__ kb,
                            const bf16_t* __restrict__ vtb,
                            const float* __restrict__ bias,
                            bf16_t* __restrict__ ob) {
  int bid = blockIdx.x;
  int it = bid & 63;
  int h  = (bid >> 6) & 7;
  int b  = bid >> 9;
  int i0 = it * 16;
  int l = threadIdx.x, m = l & 15, g = l >> 4;

  __shared__ float ldsP[16][36];  // 16 rows x 32 cols, padded

  const size_t bh = (size_t)b * H_ + h;
  const bf16_t* qrow = qb + (bh * S_ + i0 + m) * KD_;
  v16bf aq0 = load_A(qrow, g);        // kd 0..31
  v16bf aq1 = load_A(qrow + 32, g);   // kd 32..63

  v8f acc[4];
#pragma unroll
  for (int t = 0; t < 4; ++t) acc[t] = v8f_zero();
  float mrow[8], lrow[8];
#pragma unroll
  for (int r = 0; r < 8; ++r) { mrow[r] = -__builtin_inff(); lrow[r] = 0.0f; }

  const float scale = 0.125f;  // 1/sqrt(64)
  const int imax = i0 + 15;
  const float* biasbase = bias + (size_t)h * S_ * S_;

  for (int j0 = 0; j0 <= imax; j0 += 32) {
    float s[2][8];
#pragma unroll
    for (int t = 0; t < 2; ++t) {
      int jb = j0 + t * 16;
      if (jb <= imax) {  // wave-uniform
        const bf16_t* krow = kb + (bh * S_ + jb + m) * KD_;
        v16bf bk0 = *(const v16bf*)(krow + g * 16);       // kd 0..31
        v16bf bk1 = *(const v16bf*)(krow + 32 + g * 16);  // kd 32..63
        v8f cs = v8f_zero();
        cs = wmma_bf16(aq0, bk0, cs);
        cs = wmma_bf16(aq1, bk1, cs);
#pragma unroll
        for (int r = 0; r < 8; ++r)
          s[t][r] = cs[r] * scale +
                    biasbase[(size_t)(i0 + r + 8 * g) * S_ + jb + m];
      } else {
#pragma unroll
        for (int r = 0; r < 8; ++r) s[t][r] = -__builtin_inff();
      }
    }

    // online softmax update (row = r + 8*g, replicated across 16-lane group)
#pragma unroll
    for (int r = 0; r < 8; ++r) {
      float x = redmax16(fmaxf(s[0][r], s[1][r]));
      float mnew = fmaxf(mrow[r], x);
      float sc = __expf(mrow[r] - mnew);
      float p0 = __expf(s[0][r] - mnew);
      float p1 = __expf(s[1][r] - mnew);
      lrow[r] = lrow[r] * sc + redsum16(p0 + p1);
      mrow[r] = mnew;
#pragma unroll
      for (int t4 = 0; t4 < 4; ++t4) acc[t4][r] *= sc;
      int mi = r + 8 * g;
      ldsP[mi][m] = p0;
      ldsP[mi][16 + m] = p1;
    }
    __syncthreads();

    // reload P in the A-operand layout (bf16, K = 32 local j's)
    v16bf ap;
#pragma unroll
    for (int e = 0; e < 16; ++e) {
      int k = g * 16 + e;
      ap[e] = (bf16_t)ldsP[m][k];
    }
    __syncthreads();

    // P (16x32) x V^T-tile (32x16) for each of the 4 kd tiles
#pragma unroll
    for (int t4 = 0; t4 < 4; ++t4) {
      const bf16_t* vrow = vtb + (bh * KD_ + t4 * 16 + m) * S_ + j0 + g * 16;
      v16bf bv = *(const v16bf*)vrow;
      acc[t4] = wmma_bf16(ap, bv, acc[t4]);
    }
  }

  // epilogue: o = acc / l, store concat-heads layout [B,S,H*KD] bf16
#pragma unroll
  for (int t4 = 0; t4 < 4; ++t4)
#pragma unroll
    for (int r = 0; r < 8; ++r) {
      int srow = i0 + r + 8 * g;
      float o = acc[t4][r] / lrow[r];
      ob[((size_t)b * S_ + srow) * (H_ * KD_) + h * KD_ + t4 * 16 + m] =
          (bf16_t)o;
    }
}

// ---------------------------------------------------------------------------
// Output projection: out[bs, dout] = sum_din o[bs, din] * Wo[dout, din]
// One wave per 16x64 output tile.
// ---------------------------------------------------------------------------
__global__ void oproj_kernel(const bf16_t* __restrict__ ob,
                             const bf16_t* __restrict__ wo,
                             float* __restrict__ out) {
  int bid = blockIdx.x;
  int nt = bid & 7;    // D/64 = 8
  int mt = bid >> 3;   // (B*S)/16 = 512
  int l = threadIdx.x, m = l & 15, g = l >> 4;
  int row0 = mt * 16;

  const bf16_t* arow = ob + ((size_t)row0 + m) * D_;
  const bf16_t* brow_base = wo + ((size_t)nt * 64 + m) * D_;

  v8f acc[4];
#pragma unroll
  for (int t = 0; t < 4; ++t) acc[t] = v8f_zero();

  for (int kk = 0; kk < D_ / 32; ++kk) {
    int d0 = kk * 32;
    v16bf a = load_A(arow + d0, g);
#pragma unroll
    for (int t = 0; t < 4; ++t) {
      v16bf bm = *(const v16bf*)(brow_base + (size_t)(t * 16) * D_ + d0 + g * 16);
      acc[t] = wmma_bf16(a, bm, acc[t]);
    }
  }
#pragma unroll
  for (int t = 0; t < 4; ++t)
#pragma unroll
    for (int r = 0; r < 8; ++r)
      out[((size_t)row0 + r + 8 * g) * D_ + nt * 64 + t * 16 + m] = acc[t][r];
}

// ---------------------------------------------------------------------------
// Host launcher
// ---------------------------------------------------------------------------
extern "C" void kernel_launch(void* const* d_in, const int* in_sizes, int n_in,
                              void* d_out, int out_size, void* d_ws,
                              size_t ws_size, hipStream_t stream) {
  (void)in_sizes; (void)n_in; (void)out_size; (void)ws_size;

  const float* src   = (const float*)d_in[0];
  const float* Wq    = (const float*)d_in[1];
  const float* Wk    = (const float*)d_in[2];
  const float* Wv    = (const float*)d_in[3];
  const float* c_raw = (const float*)d_in[4];
  const float* Lp    = (const float*)d_in[5];
  const float* w1    = (const float*)d_in[6];
  const float* b1    = (const float*)d_in[7];
  const float* W2    = (const float*)d_in[8];
  const float* b2    = (const float*)d_in[9];
  const float* w3    = (const float*)d_in[10];
  const float* b3    = (const float*)d_in[11];
  const float* Wo    = (const float*)d_in[12];
  float* out = (float*)d_out;

  // workspace carve-up (256B aligned)
  size_t off = 0;
  auto alloc = [&](size_t bytes) -> void* {
    off = (off + 255) & ~(size_t)255;
    void* p = (char*)d_ws + off;
    off += bytes;
    return p;
  };
  const size_t nSrc = (size_t)B_ * S_ * D_;          // 4,194,304
  const size_t nW   = (size_t)H_ * KD_ * D_;         // 262,144
  const size_t nQKV = (size_t)B_ * H_ * S_ * KD_;    // 4,194,304
  const size_t nW2  = (size_t)H_ * HID_ * HID_;      // 8,192
  const size_t nWo  = (size_t)D_ * D_;               // 262,144
  const size_t nBias= (size_t)H_ * S_ * S_;          // 8,388,608

  bf16_t* src_bf = (bf16_t*)alloc(nSrc * 2);
  bf16_t* wq_bf  = (bf16_t*)alloc(nW * 2);
  bf16_t* wk_bf  = (bf16_t*)alloc(nW * 2);
  bf16_t* wv_bf  = (bf16_t*)alloc(nW * 2);
  bf16_t* w2_bf  = (bf16_t*)alloc(nW2 * 2);
  bf16_t* wo_bf  = (bf16_t*)alloc(nWo * 2);
  bf16_t* q_bf   = (bf16_t*)alloc(nQKV * 2);
  bf16_t* k_bf   = (bf16_t*)alloc(nQKV * 2);
  bf16_t* vt_bf  = (bf16_t*)alloc(nQKV * 2);
  float*  bias   = (float*)alloc(nBias * 4);
  bf16_t* o_bf   = (bf16_t*)alloc(nSrc * 2);

  auto cvt = [&](const float* in, bf16_t* o, size_t n) {
    cvt_bf16_kernel<<<(unsigned)((n + 255) / 256), 256, 0, stream>>>(in, o, (int)n);
  };
  cvt(src, src_bf, nSrc);
  cvt(Wq, wq_bf, nW);
  cvt(Wk, wk_bf, nW);
  cvt(Wv, wv_bf, nW);
  cvt(W2, w2_bf, nW2);
  cvt(Wo, wo_bf, nWo);

  // QKV projections: B*3*H*(S/16) waves
  qkv_kernel<<<B_ * 3 * H_ * (S_ / 16), 32, 0, stream>>>(
      src_bf, wq_bf, wk_bf, wv_bf, q_bf, k_bf, vt_bf);

  // FIRE bias (batch independent): H*S*(S/16) waves
  fire_bias_kernel<<<H_ * S_ * (S_ / 16), 32, 0, stream>>>(
      c_raw, Lp, w1, b1, w2_bf, b2, w3, b3, bias);

  // Flash attention: B*H*(S/16) waves
  attn_kernel<<<B_ * H_ * (S_ / 16), 32, 0, stream>>>(q_bf, k_bf, vt_bf, bias,
                                                      o_bf);

  // Output projection: (B*S/16)*(D/64) waves
  oproj_kernel<<<(B_ * S_ / 16) * (D_ / 64), 32, 0, stream>>>(o_bf, wo_bf, out);
}
